// DexWM_53626961658043
// MI455X (gfx1250) — matrix operand order repaired and checked by reading, not compiled, not run on hardware
//
#include <hip/hip_runtime.h>

typedef __attribute__((ext_vector_type(16))) __bf16 v16bf;
typedef __attribute__((ext_vector_type(8)))  float  v8f;
typedef unsigned short u16;

#define WMMA_BF16(a_, b_, c_) \
  __builtin_amdgcn_wmma_f32_16x16x32_bf16(false, (a_), false, (b_), (short)0, (c_), false, false)

// ---------------------------------------------------------------- helpers ---
__device__ __forceinline__ u16 f2b(float f) {
  unsigned u = __float_as_uint(f);
  u += 0x7fffu + ((u >> 16) & 1u);   // round-to-nearest-even
  return (u16)(u >> 16);
}

// CDNA5 async copy: 16B/lane global -> LDS, tracked by ASYNCcnt (no VGPR staging).
__device__ __forceinline__ void async_ld16(void* lds, const void* g) {
  asm volatile("global_load_async_to_lds_b128 %0, %1, off"
               :: "v"((unsigned)(unsigned long long)lds), "v"(g)
               : "memory");
}
__device__ __forceinline__ void wait_async0() {
#if __has_builtin(__builtin_amdgcn_s_wait_asynccnt)
  __builtin_amdgcn_s_wait_asynccnt(0);
#else
  asm volatile("s_wait_asynccnt 0x0" ::: "memory");
#endif
}

// A-operand fragment (16x32 bf16, ISA 7.12.2): m = lane&15,
// lanes 0-15 hold K {0..7,16..23}, lanes 16-31 hold K {8..15,24..31}, 2 K per VGPR.
__device__ __forceinline__ v16bf ldfragA(const u16* rowp, int t) {
  union { v16bf v; unsigned u[8]; } f;
#pragma unroll
  for (int j = 0; j < 8; ++j) {
    const int kk = ((j >> 2) << 4) + (t << 3) + ((j & 3) << 1);
    f.u[j] = *(const unsigned*)(rowp + kk);
  }
  return f.v;
}
// B-operand fragment (32x16 bf16): n = lane&15,
// lanes 0-15 hold K {0..7,16..23}, lanes 16-31 hold K {8..15,24..31}, 2 consecutive K per VGPR.
__device__ __forceinline__ v16bf ldfragB(const u16* rowp, int t) {
  union { v16bf v; unsigned u[8]; } f;
#pragma unroll
  for (int j = 0; j < 8; ++j) {
    const int kk = ((j >> 2) << 4) + (t << 3) + ((j & 3) << 1);
    f.u[j] = *(const unsigned*)(rowp + kk);
  }
  return f.v;
}

// ---------------------------------------------------- elementwise converts ---
__global__ __launch_bounds__(256) void cvt_bf16(const float* __restrict__ in,
                                                u16* __restrict__ out, int n) {
  for (int i = blockIdx.x * 256 + threadIdx.x; i < n; i += gridDim.x * 256)
    out[i] = f2b(in[i]);
}
__global__ __launch_bounds__(256) void silu_bf16(const float* __restrict__ in,
                                                 u16* __restrict__ out, int n) {
  for (int i = blockIdx.x * 256 + threadIdx.x; i < n; i += gridDim.x * 256) {
    float v = in[i];
    out[i] = f2b(v / (1.0f + __expf(-v)));
  }
}

// --------------------------------------------- LayerNorm + adaLN modulate ---
__global__ __launch_bounds__(256) void ln_mod(const float* __restrict__ X,
                                              const float* __restrict__ ada,
                                              int shift_ofs, int scale_ofs,
                                              u16* __restrict__ out) {
  const int row  = blockIdx.x;
  const int tid  = threadIdx.x;
  const int lane = tid & 31, wave = tid >> 5;
  __shared__ float sh[16];
  const float* xr = X + (size_t)row * 1024;
  float v[4], s = 0.f, s2 = 0.f;
#pragma unroll
  for (int j = 0; j < 4; ++j) {
    v[j] = xr[tid + j * 256];
    s += v[j]; s2 += v[j] * v[j];
  }
#pragma unroll
  for (int d = 16; d > 0; d >>= 1) { s += __shfl_xor(s, d, 32); s2 += __shfl_xor(s2, d, 32); }
  if (lane == 0) { sh[wave] = s; sh[wave + 8] = s2; }
  __syncthreads();
  if (tid == 0) {
    float ts = 0.f, ts2 = 0.f;
    for (int w = 0; w < 8; ++w) { ts += sh[w]; ts2 += sh[w + 8]; }
    sh[0] = ts * (1.f / 1024.f);
    sh[1] = ts2 * (1.f / 1024.f);
  }
  __syncthreads();
  const float mu = sh[0];
  const float rstd = rsqrtf(sh[1] - mu * mu + 1e-6f);
  const float* ap = ada + (size_t)(row / 196) * 11264;
#pragma unroll
  for (int j = 0; j < 4; ++j) {
    const int col = tid + j * 256;
    const float y = (v[j] - mu) * rstd * (1.f + ap[scale_ofs * 1024 + col]) + ap[shift_ofs * 1024 + col];
    out[(size_t)row * 1024 + col] = f2b(y);
  }
}

// -------------------------------------------------------- WMMA tiled GEMM ---
// C(MxN) = A(MxK bf16) x B(NxK bf16)^T.  Block tile 128x128, 8 waves 4x2,
// wave tile 32x64 (8 wmma / K-step).  Double-buffered async-to-LDS pipeline.
#define GBM 128
#define GBN 128
#define LDSA 40   // 32 + 8 pad (ushorts): conflict-free fragment reads, 16B-aligned rows
__global__ __launch_bounds__(256) void wgemm(const u16* __restrict__ A,
                                             const u16* __restrict__ B,
                                             const float* __restrict__ bias,
                                             int M, int N, int K,
                                             int epi,          // 0:f32 1:bf16 2:gelu-bf16 3:gated-resid
                                             float* __restrict__ outF,
                                             u16* __restrict__ outH,
                                             const float* __restrict__ adaGate,
                                             int gofs) {
  __shared__ u16 As[2][GBM * LDSA];
  __shared__ u16 Bs[2][GBN * LDSA];

  const int tid  = threadIdx.x;
  const int lane = tid & 31, wave = tid >> 5;
  const int t = lane >> 4, ln = lane & 15;
  const int wm = wave & 3, wn = wave >> 2;
  const int bm = blockIdx.y * GBM;
  const int bn = blockIdx.x * GBN;

  v8f acc[2][4];
#pragma unroll
  for (int u = 0; u < 2; ++u)
#pragma unroll
    for (int s = 0; s < 4; ++s)
#pragma unroll
      for (int r = 0; r < 8; ++r) acc[u][s][r] = 0.f;

  // staging map: 128 rows x 32 cols per operand, 2x16B per thread per operand
  const int srow = tid >> 1;               // 0..127
  const int scol = (tid & 1) << 4;         // 0 | 16 (elements)
  const int arowc = (bm + srow < M) ? (bm + srow) : (M - 1);   // clamp (EXEC stays full)
  const size_t aoff = (size_t)arowc * K;
  const size_t boff = (size_t)(bn + srow) * K;

  auto stage = [&](int buf, int kb) {
    async_ld16(&As[buf][srow * LDSA + scol],     A + aoff + kb + scol);
    async_ld16(&As[buf][srow * LDSA + scol + 8], A + aoff + kb + scol + 8);
    async_ld16(&Bs[buf][srow * LDSA + scol],     B + boff + kb + scol);
    async_ld16(&Bs[buf][srow * LDSA + scol + 8], B + boff + kb + scol + 8);
  };

  stage(0, 0);
  wait_async0();
  __syncthreads();

  int buf = 0;
  for (int kb = 0; kb < K; kb += 32) {
    if (kb + 32 < K) stage(buf ^ 1, kb + 32);
    if (kb + 64 < K) {                        // L2 prefetch two slabs ahead
      __builtin_prefetch(A + aoff + kb + 64 + scol, 0, 1);
      __builtin_prefetch(B + boff + kb + 64 + scol, 0, 1);
    }
    const v16bf a0 = ldfragA(&As[buf][(wm * 32 + ln) * LDSA], t);
    const v16bf a1 = ldfragA(&As[buf][(wm * 32 + 16 + ln) * LDSA], t);
#pragma unroll
    for (int s = 0; s < 4; ++s) {
      const v16bf bfm = ldfragB(&Bs[buf][(wn * 64 + s * 16 + ln) * LDSA], t);
      acc[0][s] = WMMA_BF16(a0, bfm, acc[0][s]);
      acc[1][s] = WMMA_BF16(a1, bfm, acc[1][s]);
    }
    wait_async0();
    __syncthreads();
    buf ^= 1;
  }

#pragma unroll
  for (int u = 0; u < 2; ++u)
#pragma unroll
    for (int s = 0; s < 4; ++s) {
      const int gcol = bn + wn * 64 + s * 16 + ln;
      const float bb = bias[gcol];
#pragma unroll
      for (int r = 0; r < 8; ++r) {
        const int grow = bm + wm * 32 + u * 16 + r + (t << 3);
        if (grow >= M) continue;
        float v = acc[u][s][r] + bb;
        if (epi == 0) {
          outF[(size_t)grow * N + gcol] = v;
        } else if (epi == 1) {
          outH[(size_t)grow * N + gcol] = f2b(v);
        } else if (epi == 2) {                   // tanh-GELU
          const float v3 = v * v * v;
          v = 0.5f * v * (1.f + tanhf(0.7978845608f * (v + 0.044715f * v3)));
          outH[(size_t)grow * N + gcol] = f2b(v);
        } else {                                 // x += gate * (o + bias)
          const float g = adaGate[(size_t)(grow / 196) * 11264 + gofs * 1024 + gcol];
          outF[(size_t)grow * N + gcol] += g * v;
        }
      }
    }
}

// ------------------------------------------------- unified flash attention ---
// mode 0: spatial (1 chunk of 196 keys); mode 1: temporal (frames < f; f==0 -> 0)
// grid = (13 q-tiles, B*F, H); dh=64, scale=0.125
#define AKW 224
#define KLD 72
#define VLD 232
__global__ __launch_bounds__(256) void attn(const u16* __restrict__ qbuf, int qs,
                                            const u16* __restrict__ kvbuf, int ks,
                                            int ko, int vo,
                                            u16* __restrict__ obuf, int mode) {
  __shared__ u16   KV[AKW * KLD];           // union: K chunk (224x64) then V^T (64x224)
  __shared__ u16   Qs[16 * KLD];
  __shared__ float Ssm[16][AKW];
  __shared__ u16   Psm[16][AKW];
  __shared__ float row_m[16], row_l[16], row_f[16];

  const int qt = blockIdx.x, y = blockIdx.y, h = blockIdx.z;
  const int tid = threadIdx.x;
  const int lane = tid & 31, wave = tid >> 5;
  const int t = lane >> 4, ln = lane & 15;
  const int q_row0 = y * 196;
  const int qo = h * 64;

  int kv_row0, nch;
  if (mode == 0) { kv_row0 = q_row0; nch = 1; }
  else {
    const int f = y % 5;
    kv_row0 = (y - f) * 196;
    nch = f;
    if (nch == 0) {                                  // frame 0: empty mask -> zero
      for (int idx = tid; idx < 16 * 64; idx += 256) {
        const int r = idx >> 6, d = idx & 63;
        const int qr = qt * 16 + r;
        if (qr < 196) obuf[(size_t)(q_row0 + qr) * 1024 + qo + d] = 0;
      }
      return;
    }
  }

  // stage Q tile (async, rows clamped; pad rows are masked out of stores later)
  if (tid < 128) {
    const int r = tid >> 3, d8 = (tid & 7) << 3;
    const int qr = (qt * 16 + r < 196) ? (qt * 16 + r) : 195;
    async_ld16(&Qs[r * KLD + d8], qbuf + (size_t)(q_row0 + qr) * qs + qo + d8);
  }
  if (tid < 16) { row_m[tid] = -3.0e38f; row_l[tid] = 0.f; }

  v8f oacc = {0, 0, 0, 0, 0, 0, 0, 0};

  for (int ch = 0; ch < nch; ++ch) {
    __syncthreads();                                        // KV buffer reuse
    for (int idx = tid; idx < AKW * 8; idx += 256) {        // stage K chunk (async)
      const int tk = idx >> 3, d8 = (idx & 7) << 3;
      const int kr = (tk < 196) ? tk : 195;                 // clamp; cols>=196 masked
      async_ld16(&KV[tk * KLD + d8],
                 kvbuf + (size_t)(kv_row0 + ch * 196 + kr) * ks + ko + qo + d8);
    }
    wait_async0();
    __syncthreads();
    for (int nt = wave; nt < 14; nt += 8) {                 // S = Q K^T (scaled)
      v8f sacc = {0, 0, 0, 0, 0, 0, 0, 0};
#pragma unroll
      for (int kk = 0; kk < 64; kk += 32) {
        const v16bf a = ldfragA(Qs + ln * KLD + kk, t);
        const v16bf b = ldfragB(KV + (nt * 16 + ln) * KLD + kk, t);
        sacc = WMMA_BF16(a, b, sacc);
      }
#pragma unroll
      for (int r = 0; r < 8; ++r)
        Ssm[r + (t << 3)][nt * 16 + ln] = sacc[r] * 0.125f;
    }
    __syncthreads();
    {                                                       // online softmax (chunk)
      const int m = tid >> 4, i = tid & 15;
      float cmax = -3.0e38f;
      for (int cc = i; cc < 196; cc += 16) cmax = fmaxf(cmax, Ssm[m][cc]);
#pragma unroll
      for (int d = 1; d < 16; d <<= 1) cmax = fmaxf(cmax, __shfl_xor(cmax, d, 32));
      const float oldm = row_m[m];
      const float newm = fmaxf(oldm, cmax);
      float csum = 0.f;
      for (int cc = i; cc < AKW; cc += 16) {
        const float e = (cc < 196) ? __expf(Ssm[m][cc] - newm) : 0.f;
        Psm[m][cc] = f2b(e);
        csum += e;
      }
#pragma unroll
      for (int d = 1; d < 16; d <<= 1) csum += __shfl_xor(csum, d, 32);
      if (i == 0) {
        const float fr = __expf(oldm - newm);
        row_f[m] = fr;
        row_l[m] = row_l[m] * fr + csum;
        row_m[m] = newm;
      }
    }
    __syncthreads();
    for (int idx = tid; idx < 64 * AKW; idx += 256) {       // stage V^T (transpose)
      const int d = idx / AKW, tk = idx % AKW;
      u16 v = 0;
      if (tk < 196) v = kvbuf[(size_t)(kv_row0 + ch * 196 + tk) * ks + vo + qo + d];
      KV[d * VLD + tk] = v;
    }
    __syncthreads();
    if (wave < 4) {                                         // O += P V (wave w: dh cols w*16..)
#pragma unroll
      for (int r = 0; r < 8; ++r) oacc[r] *= row_f[r + (t << 3)];
      for (int kk = 0; kk < AKW; kk += 32) {
        const v16bf a = ldfragA(&Psm[0][0] + ln * AKW + kk, t);
        const v16bf b = ldfragB(KV + (wave * 16 + ln) * VLD + kk, t);
        oacc = WMMA_BF16(a, b, oacc);
      }
    }
  }
  if (wave < 4) {
#pragma unroll
    for (int r = 0; r < 8; ++r) {
      const int m = r + (t << 3);
      const int qr = qt * 16 + m;
      if (qr < 196) {
        const float o = oacc[r] / fmaxf(row_l[m], 1e-30f);
        obuf[(size_t)(q_row0 + qr) * 1024 + qo + wave * 16 + ln] = f2b(o);
      }
    }
  }
}

// --------------------------------------------------------------- launcher ---
extern "C" void kernel_launch(void* const* d_in, const int* in_sizes, int n_in,
                              void* d_out, int out_size, void* d_ws, size_t ws_size,
                              hipStream_t stream) {
  const float* x      = (const float*)d_in[0];
  const float* c      = (const float*)d_in[1];
  const float* xclean = (const float*)d_in[2];
  const float* W_ada  = (const float*)d_in[3];
  const float* b_ada  = (const float*)d_in[4];
  const float* Wqkv_s = (const float*)d_in[5];
  const float* bqkv_s = (const float*)d_in[6];
  const float* Wo_s   = (const float*)d_in[7];
  const float* bo_s   = (const float*)d_in[8];
  const float* Wqkv_t = (const float*)d_in[9];
  const float* bqkv_t = (const float*)d_in[10];
  const float* Wo_t   = (const float*)d_in[11];
  const float* bo_t   = (const float*)d_in[12];
  const float* W1     = (const float*)d_in[13];
  const float* b1     = (const float*)d_in[14];
  const float* W2     = (const float*)d_in[15];
  const float* b2     = (const float*)d_in[16];
  (void)in_sizes; (void)n_in; (void)out_size; (void)ws_size;

  const int Mr = 7840;                         // B*F*N token rows
  float* xout = (float*)d_out;                 // running residual x
  char* ws = (char*)d_ws;

  size_t o = 0;
  u16* wada_b   = (u16*)(ws + o); o += 11264ull * 1024 * 2;
  u16* wqkvs_b  = (u16*)(ws + o); o += 3072ull  * 1024 * 2;
  u16* wos_b    = (u16*)(ws + o); o += 1024ull  * 1024 * 2;
  u16* wqkvt_b  = (u16*)(ws + o); o += 3072ull  * 1024 * 2;
  u16* wot_b    = (u16*)(ws + o); o += 1024ull  * 1024 * 2;
  u16* w1_b     = (u16*)(ws + o); o += 2048ull  * 1024 * 2;
  u16* w2_b     = (u16*)(ws + o); o += 1024ull  * 2048 * 2;
  u16* siluc_b  = (u16*)(ws + o); o += 40ull    * 1024 * 2;
  float* adaf   = (float*)(ws + o); o += 40ull  * 11264 * 4;
  u16* xa       = (u16*)(ws + o); o += (size_t)Mr * 1024 * 2;
  u16* xb       = (u16*)(ws + o); o += (size_t)Mr * 1024 * 2;
  u16* big      = (u16*)(ws + o); o += (size_t)Mr * 3072 * 2;   // qkv_s | q_t+kv_t | mlp-h
  u16* attno    = (u16*)(ws + o); o += (size_t)Mr * 1024 * 2;
  u16* qt_b  = big;
  u16* kvt_b = big + (size_t)Mr * 1024;
  u16* h_b   = big;

  // weights -> bf16
  cvt_bf16<<<4096, 256, 0, stream>>>(W_ada,  wada_b,  11264 * 1024);
  cvt_bf16<<<2048, 256, 0, stream>>>(Wqkv_s, wqkvs_b, 3072 * 1024);
  cvt_bf16<<<1024, 256, 0, stream>>>(Wo_s,   wos_b,   1024 * 1024);
  cvt_bf16<<<2048, 256, 0, stream>>>(Wqkv_t, wqkvt_b, 3072 * 1024);
  cvt_bf16<<<1024, 256, 0, stream>>>(Wo_t,   wot_b,   1024 * 1024);
  cvt_bf16<<<1024, 256, 0, stream>>>(W1,     w1_b,    2048 * 1024);
  cvt_bf16<<<1024, 256, 0, stream>>>(W2,     w2_b,    1024 * 2048);

  hipMemcpyAsync(xout, x, (size_t)Mr * 1024 * 4, hipMemcpyDeviceToDevice, stream);

  // adaLN: ada = silu(c) @ W_ada^T + b_ada        (40 x 11264)
  silu_bf16<<<160, 256, 0, stream>>>(c, siluc_b, 40 * 1024);
  wgemm<<<dim3(11264 / GBN, 1), 256, 0, stream>>>(siluc_b, wada_b, b_ada, 40, 11264, 1024,
                                                  0, adaf, nullptr, nullptr, 0);

  // ---- spatial self-attention ----
  ln_mod<<<Mr, 256, 0, stream>>>(xout, adaf, 0, 1, xa);
  wgemm<<<dim3(3072 / GBN, 62), 256, 0, stream>>>(xa, wqkvs_b, bqkv_s, Mr, 3072, 1024,
                                                  1, nullptr, big, nullptr, 0);
  attn<<<dim3(13, 40, 16), 256, 0, stream>>>(big, 3072, big, 3072, 1024, 2048, attno, 0);
  wgemm<<<dim3(1024 / GBN, 62), 256, 0, stream>>>(attno, wos_b, bo_s, Mr, 1024, 1024,
                                                  3, xout, nullptr, adaf, 2);

  // ---- temporal cross-attention ----
  ln_mod<<<Mr, 256, 0, stream>>>(xclean, adaf, 3, 4, xb);    // xcn
  ln_mod<<<Mr, 256, 0, stream>>>(xout,   adaf, 5, 6, xa);    // xn2
  wgemm<<<dim3(1024 / GBN, 62), 256, 0, stream>>>(xa, wqkvt_b, bqkv_t, Mr, 1024, 1024,
                                                  1, nullptr, qt_b, nullptr, 0);
  wgemm<<<dim3(2048 / GBN, 62), 256, 0, stream>>>(xb, wqkvt_b + 1024ull * 1024, bqkv_t + 1024,
                                                  Mr, 2048, 1024, 1, nullptr, kvt_b, nullptr, 0);
  attn<<<dim3(13, 40, 16), 256, 0, stream>>>(qt_b, 1024, kvt_b, 2048, 0, 1024, attno, 1);
  wgemm<<<dim3(1024 / GBN, 62), 256, 0, stream>>>(attno, wot_b, bo_t, Mr, 1024, 1024,
                                                  3, xout, nullptr, adaf, 7);

  // ---- MLP ----
  ln_mod<<<Mr, 256, 0, stream>>>(xout, adaf, 8, 9, xa);      // xn3
  wgemm<<<dim3(2048 / GBN, 62), 256, 0, stream>>>(xa, w1_b, b1, Mr, 2048, 1024,
                                                  2, nullptr, h_b, nullptr, 0);
  wgemm<<<dim3(1024 / GBN, 62), 256, 0, stream>>>(h_b, w2_b, b2, Mr, 1024, 2048,
                                                  3, xout, nullptr, adaf, 10);
}
